// GraphEncoder_6983616823298
// MI455X (gfx1250) — matrix operand order repaired and checked by reading, not compiled, not run on hardware
//
#include <hip/hip_runtime.h>
#include <hip/hip_bf16.h>

#define NODES 50000
#define EDGES 800000
#define F_IN  11
#define HID   128
#define HEADS 4
#define CPH   32
#define LAYERS 4
#define NEG_SLOPE 0.2f
#define NEG_INF (-__builtin_huge_valf())

typedef __attribute__((ext_vector_type(16))) _Float16 v16h;
typedef __attribute__((ext_vector_type(8)))  float    v8f;

// ---------------- weight convert: fp32 -> fp16 (row-major [o,k] kept) ----------------
__global__ void k_convert_weights(const float* __restrict__ Wl, const float* __restrict__ Wr,
                                  _Float16* __restrict__ Wlh, _Float16* __restrict__ Wrh) {
    int i = blockIdx.x * blockDim.x + threadIdx.x;      // LAYERS*HID*HID = 65536
    if (i < LAYERS * HID * HID) {
        Wlh[i] = (_Float16)Wl[i];
        Wrh[i] = (_Float16)Wr[i];
    }
}

// ---------------- CSR build ----------------
__global__ void k_zero_ints(int* __restrict__ p, int n) {
    int i = blockIdx.x * blockDim.x + threadIdx.x;
    if (i < n) p[i] = 0;
}

__global__ void k_count_deg(const int* __restrict__ dst, int* __restrict__ counts) {
    int e = blockIdx.x * blockDim.x + threadIdx.x;
    if (e < EDGES) atomicAdd(&counts[dst[e]], 1);
}

// single-workgroup chunked Hillis-Steele exclusive scan (N=50000 -> 49 chunks)
__global__ void k_scan(const int* __restrict__ counts, int* __restrict__ rowstart) {
    __shared__ int buf[1024];
    __shared__ int carry;
    int t = threadIdx.x;
    if (t == 0) carry = 0;
    __syncthreads();
    for (int base = 0; base < NODES; base += 1024) {
        int i = base + t;
        int c = (i < NODES) ? counts[i] : 0;
        buf[t] = c;
        __syncthreads();
        for (int off = 1; off < 1024; off <<= 1) {
            int v = (t >= off) ? buf[t - off] : 0;
            __syncthreads();
            buf[t] += v;
            __syncthreads();
        }
        int excl = carry + buf[t] - c;
        if (i < NODES) rowstart[i] = excl;
        __syncthreads();                       // everyone read carry
        if (t == 1023) carry += buf[1023];
        __syncthreads();
    }
    if (t == 0) rowstart[NODES] = carry;       // == EDGES
}

__global__ void k_fill_csr(const int* __restrict__ dst, const int* __restrict__ rowstart,
                           int* __restrict__ cursor, int* __restrict__ eids) {
    int e = blockIdx.x * blockDim.x + threadIdx.x;
    if (e >= EDGES) return;
    int d = dst[e];
    int pos = rowstart[d] + atomicAdd(&cursor[d], 1);
    eids[pos] = e;
}

// ---------------- embedding: x = relu(NF @ W_emb^T + b_emb), K=11 (scalar) ----------------
__global__ void k_embed(const float* __restrict__ nf, const float* __restrict__ W,
                        const float* __restrict__ b, float* __restrict__ x) {
    int t = blockIdx.x * blockDim.x + threadIdx.x;      // NODES*HID
    if (t >= NODES * HID) return;
    int n = t >> 7, o = t & 127;
    const float* r = nf + n * F_IN;
    const float* w = W + o * F_IN;
    float s = b[o];
    #pragma unroll
    for (int k = 0; k < F_IN; ++k) s += r[k] * w[k];
    x[t] = fmaxf(s, 0.f);
}

// ---------------- WMMA GEMM: xl = x@Wl^T + bl ; xr = x@Wr^T + br ----------------
// Block = 256 threads = 8 waves; block handles 16 rows x 128 cols (full HID).
// The shared 16x128 fp32 x-tile is staged ONCE per block with async global->LDS
// (ASYNCcnt path), then repacked fp32->fp16 into per-lane A-fragment order (contiguous k) so
// each lane's per-chunk A fragment is a single aligned 32B LDS read.
__global__ void k_gemm_xlxr(const float* __restrict__ x,
                            const _Float16* __restrict__ Wlh, const _Float16* __restrict__ Wrh,
                            const float* __restrict__ bl, const float* __restrict__ br,
                            float* __restrict__ xl, float* __restrict__ xr) {
    __shared__ __align__(16) float    sA[16 * 128];        // raw fp32 tile (8KB)
    __shared__ __align__(32) _Float16 sP[4][2][16][16];    // A-frags: [kc][laneHalf][m][16 halves] (4KB)

    const int t    = threadIdx.x;
    const int wave = t >> 5;                   // 0..7 -> column tile
    const int lane = t & 31;
    const int mt   = blockIdx.x;               // 0..3124 (50000/16)
    const int mrow = lane & 15;                // A row within tile
    const int hs   = lane >> 4;                // lane half-group
    const int col  = wave * 16 + (lane & 15);  // B/D column (output channel)

    // ---- async stage: 8KB contiguous tile -> LDS (2 x b128 per lane) ----
    {
        const char* gsrc = (const char*)(x + (size_t)mt * 16 * HID);
        unsigned lds0 = (unsigned)(uintptr_t)(&sA[0]) + (unsigned)t * 16u;
        unsigned long long g0 = (unsigned long long)(uintptr_t)gsrc + (unsigned long long)t * 16ull;
        asm volatile("global_load_async_to_lds_b128 %0, %1, off"
                     :: "v"(lds0), "v"(g0) : "memory");
        asm volatile("global_load_async_to_lds_b128 %0, %1, off"
                     :: "v"(lds0 + 4096u), "v"(g0 + 4096ull) : "memory");
        asm volatile("s_wait_asynccnt 0x0" ::: "memory");
    }
    __syncthreads();

    // ---- repack fp32 -> fp16 in A-fragment order, contiguous k = 16*half + s (same as B) ----
    if (t < 128) {                             // one group per (kc, hs, m)
        int m = t & 15, h2 = (t >> 4) & 1, kc = t >> 5;
        int k0 = kc * 32;
        #pragma unroll
        for (int j = 0; j < 8; ++j) {
            int kb = h2 * 16 + 2 * j;
            sP[kc][h2][m][2 * j]     = (_Float16)sA[m * 128 + k0 + kb];
            sP[kc][h2][m][2 * j + 1] = (_Float16)sA[m * 128 + k0 + kb + 1];
        }
    }
    __syncthreads();

    v8f accL = {}; v8f accR = {};
    #pragma unroll
    for (int kc = 0; kc < 4; ++kc) {           // K = 128 in chunks of 32
        int k0 = kc * 32;
        const v16h a  = *(const v16h*)(&sP[kc][hs][mrow][0]);   // 32B LDS read
        // B frag: lanes 0-15 K=k0..k0+15, lanes 16-31 K=k0+16..k0+31; contiguous k
        const v16h bL = *(const v16h*)(Wlh + col * HID + k0 + hs * 16);
        const v16h bR = *(const v16h*)(Wrh + col * HID + k0 + hs * 16);
        accL = __builtin_amdgcn_wmma_f32_16x16x32_f16(false, a, false, bL, (short)0, accL, false, false);
        accR = __builtin_amdgcn_wmma_f32_16x16x32_f16(false, a, false, bR, (short)0, accR, false, false);
    }

    // D layout: VGPR r -> row = r + 8*laneHalf, col = lane&15 (within tile)
    float vbl = bl[col], vbr = br[col];
    #pragma unroll
    for (int r = 0; r < 8; ++r) {
        int row = mt * 16 + r + hs * 8;
        xl[row * HID + col] = accL[r] + vbl;
        xr[row * HID + col] = accR[r] + vbr;
    }
}

// ---------------- edge logits: logits[e,h] = sum_c lrelu(xl[src]+xr[dst]) * att[h,c] ------
__global__ void k_edge_logits(const float* __restrict__ xl, const float* __restrict__ xr,
                              const int* __restrict__ src, const int* __restrict__ dst,
                              const float* __restrict__ att, float* __restrict__ logits) {
    int t = blockIdx.x * blockDim.x + threadIdx.x;      // EDGES*HEADS
    if (t >= EDGES * HEADS) return;
    int e = t >> 2, h = t & 3;
    const float4* a = (const float4*)(xl + src[e] * HID + h * CPH);
    const float4* b = (const float4*)(xr + dst[e] * HID + h * CPH);
    const float4* w = (const float4*)(att + h * CPH);
    float s = 0.f;
    #pragma unroll
    for (int i = 0; i < CPH / 4; ++i) {
        float4 u = a[i], v = b[i], ww = w[i];
        float z;
        z = u.x + v.x; z = z > 0.f ? z : NEG_SLOPE * z; s += z * ww.x;
        z = u.y + v.y; z = z > 0.f ? z : NEG_SLOPE * z; s += z * ww.y;
        z = u.z + v.z; z = z > 0.f ? z : NEG_SLOPE * z; s += z * ww.z;
        z = u.w + v.w; z = z > 0.f ? z : NEG_SLOPE * z; s += z * ww.w;
    }
    logits[t] = s;
}

// ---------------- node-parallel segment softmax + aggregate + epilogue ----------------
// one wave per (node, head); lane = channel. No float atomics anywhere.
__global__ void k_node_aggregate(const float* __restrict__ xl, const float* __restrict__ logits,
                                 const int* __restrict__ rowstart, const int* __restrict__ eids,
                                 const int* __restrict__ src, const float* __restrict__ biasL,
                                 const float* __restrict__ xres, float* __restrict__ xout,
                                 int layer) {
    int gid  = blockIdx.x * blockDim.x + threadIdx.x;
    int wid  = gid >> 5;                         // (node, head) index
    int lane = gid & 31;                         // channel
    if (wid >= NODES * HEADS) return;
    int n = wid >> 2, h = wid & 3;
    int rs = rowstart[n], re = rowstart[n + 1];

    float m = NEG_INF;
    for (int i = rs; i < re; ++i)                // uniform (broadcast) loads
        m = fmaxf(m, logits[eids[i] * HEADS + h]);

    float denom = 0.f, acc = 0.f;
    if (re > rs) {
        int e = eids[rs];
        const float* rowp = xl + src[e] * HID + h * CPH;
        for (int i = rs; i < re; ++i) {
            const float* cur = rowp;
            float lg = logits[e * HEADS + h];
            if (i + 1 < re) {                    // prefetch next source row (global_prefetch_b8)
                e = eids[i + 1];
                rowp = xl + src[e] * HID + h * CPH;
                __builtin_prefetch(rowp, 0, 0);
            }
            float p = __expf(lg - m);
            denom += p;
            acc += p * cur[lane];                // coalesced 128B/edge
        }
    }
    float v = (re > rs) ? (acc / denom) : 0.f;
    v = fmaxf(v + biasL[h * CPH + lane], 0.f);          // +bias, relu
    if (layer > 0) v += xres[n * HID + h * CPH + lane]; // residual
    xout[n * HID + h * CPH + lane] = v;
}

// ---------------- graph mean/max over nodes ----------------
__global__ void k_graph_reduce(const float* __restrict__ x, float* __restrict__ out) {
    __shared__ float ssum[256];
    __shared__ float smax[256];
    int c = blockIdx.x, t = threadIdx.x;
    float s = 0.f, mx = NEG_INF;
    for (int n = t; n < NODES; n += 256) {
        float v = x[n * HID + c];
        s += v; mx = fmaxf(mx, v);
    }
    ssum[t] = s; smax[t] = mx;
    __syncthreads();
    for (int off = 128; off > 0; off >>= 1) {
        if (t < off) { ssum[t] += ssum[t + off]; smax[t] = fmaxf(smax[t], smax[t + off]); }
        __syncthreads();
    }
    if (t == 0) { out[c] = ssum[0] / (float)NODES; out[HID + c] = smax[0]; }
}

__global__ void k_copy_out(const float* __restrict__ x, float* __restrict__ out) {
    int i = blockIdx.x * blockDim.x + threadIdx.x;
    if (i < NODES * HID) out[i] = x[i];
}

// ---------------- host orchestration ----------------
extern "C" void kernel_launch(void* const* d_in, const int* in_sizes, int n_in,
                              void* d_out, int out_size, void* d_ws, size_t ws_size,
                              hipStream_t stream) {
    const float* nf    = (const float*)d_in[0];
    const int*   eidx  = (const int*)d_in[1];
    const float* W_emb = (const float*)d_in[3];
    const float* b_emb = (const float*)d_in[4];
    const float* Wl    = (const float*)d_in[5];
    const float* bl    = (const float*)d_in[6];
    const float* Wr    = (const float*)d_in[7];
    const float* br    = (const float*)d_in[8];
    const float* att   = (const float*)d_in[9];
    const float* bias  = (const float*)d_in[10];
    float* out = (float*)d_out;

    const int* src = eidx;
    const int* dst = eidx + EDGES;

    // workspace carve-up (256B aligned)
    auto align = [](size_t v) { return (v + 255) & ~(size_t)255; };
    char* ws = (char*)d_ws;
    size_t off = 0;
    float* x_a    = (float*)(ws + off); off = align(off + (size_t)NODES * HID * 4);
    float* x_b    = (float*)(ws + off); off = align(off + (size_t)NODES * HID * 4);
    float* xl     = (float*)(ws + off); off = align(off + (size_t)NODES * HID * 4);
    float* xr     = (float*)(ws + off); off = align(off + (size_t)NODES * HID * 4);
    float* logits = (float*)(ws + off); off = align(off + (size_t)EDGES * HEADS * 4);
    _Float16* Wlh = (_Float16*)(ws + off); off = align(off + (size_t)LAYERS * HID * HID * 2);
    _Float16* Wrh = (_Float16*)(ws + off); off = align(off + (size_t)LAYERS * HID * HID * 2);
    int* counts   = (int*)(ws + off); off = align(off + (size_t)NODES * 4);
    int* cursor   = (int*)(ws + off); off = align(off + (size_t)NODES * 4);
    int* rowstart = (int*)(ws + off); off = align(off + (size_t)(NODES + 1) * 4);
    int* eids     = (int*)(ws + off); off = align(off + (size_t)EDGES * 4);

    // weights fp32 -> fp16 (once per call)
    k_convert_weights<<<(LAYERS * HID * HID + 255) / 256, 256, 0, stream>>>(Wl, Wr, Wlh, Wrh);

    // CSR by destination (dst fixed across layers)
    k_zero_ints<<<(2 * NODES + 255) / 256, 256, 0, stream>>>(counts, 2 * NODES); // counts, 192B pad, cursor; last 48 cursor ints not zeroed
    k_count_deg<<<(EDGES + 255) / 256, 256, 0, stream>>>(dst, counts);
    k_scan<<<1, 1024, 0, stream>>>(counts, rowstart);
    k_fill_csr<<<(EDGES + 255) / 256, 256, 0, stream>>>(dst, rowstart, cursor, eids);

    // embedding
    k_embed<<<(NODES * HID + 255) / 256, 256, 0, stream>>>(nf, W_emb, b_emb, x_a);

    float* x_cur = x_a;
    float* x_nxt = x_b;
    for (int layer = 0; layer < LAYERS; ++layer) {
        k_gemm_xlxr<<<NODES / 16, 256, 0, stream>>>(
            x_cur, Wlh + (size_t)layer * HID * HID, Wrh + (size_t)layer * HID * HID,
            bl + layer * HID, br + layer * HID, xl, xr);
        k_edge_logits<<<(EDGES * HEADS + 255) / 256, 256, 0, stream>>>(
            xl, xr, src, dst, att + layer * HEADS * CPH, logits);
        k_node_aggregate<<<(NODES * HEADS * 32 + 255) / 256, 256, 0, stream>>>(
            xl, logits, rowstart, eids, src, bias + layer * HID, x_cur, x_nxt, layer);
        float* tmp = x_cur; x_cur = x_nxt; x_nxt = tmp;
    }

    k_graph_reduce<<<HID, 256, 0, stream>>>(x_cur, out);
    k_copy_out<<<(NODES * HID + 255) / 256, 256, 0, stream>>>(x_cur, out + 2 * HID);
}
